// MoE_90640989815287
// MI455X (gfx1250) — matrix operand order repaired and checked by reading, not compiled, not run on hardware
//
#include <hip/hip_runtime.h>
#include <hip/hip_bf16.h>
#include <stdint.h>

typedef uint16_t u16;
typedef __attribute__((ext_vector_type(16))) __bf16 v16bf;
typedef __attribute__((ext_vector_type(8)))  float  v8f;

static constexpr int Dd   = 2048;
static constexpr int Hh   = 1408;
static constexpr int Ee   = 16;
static constexpr int NTOK = 8192;
static constexpr int KSEL = 4;
static constexpr int SHd  = 2816;

static constexpr int BM  = 128;
static constexpr int BN  = 128;
static constexpr int BK  = 32;
static constexpr int LDT = 40;                       // padded LDS row stride (u16): 80B, conflict-free
static constexpr uint32_t TILE_BYTES = BM * LDT * 2; // 10240 B per tile buffer
static constexpr int      TILE_ELEMS = BM * LDT;

// ---------------- small helpers ----------------

__device__ __forceinline__ u16 f2bf(float f) {
    uint32_t u = __float_as_uint(f);
    u += 0x7FFFu + ((u >> 16) & 1u);     // round-to-nearest-even
    return (u16)(u >> 16);
}

union FragU { uint4 u[2]; v16bf v; };

// A fragment (16x32 bf16): lane L holds row M=L&15; K chunks [c0..c0+7],[c0+16..c0+23], c0=8*(L>=16)
__device__ __forceinline__ v16bf load_a(const u16* As, int mBase, int lane) {
    const int l15 = lane & 15, lhi = (lane >> 4) & 1;
    const u16* p = As + (size_t)(mBase + l15) * LDT + lhi * 8;
    FragU f;
    f.u[0] = *(const uint4*)(p);
    f.u[1] = *(const uint4*)(p + 16);
    return f.v;
}

// B fragment (32x16 bf16), B stored column-major in LDS as [n][k] rows:
// lane L holds col N=L&15, K chunks [c0..c0+7],[c0+16..c0+23], c0=8*lhi
__device__ __forceinline__ v16bf load_b(const u16* Bs, int nBase, int lane) {
    const int l15 = lane & 15, lhi = (lane >> 4) & 1;
    const u16* p = Bs + (size_t)(nBase + l15) * LDT + lhi * 8;
    FragU f;
    f.u[0] = *(const uint4*)(p);
    f.u[1] = *(const uint4*)(p + 16);
    return f.v;
}

__device__ __forceinline__ v8f wmma_bf16(v16bf a, v16bf b, v8f c) {
    return __builtin_amdgcn_wmma_f32_16x16x32_bf16(false, a, false, b, (short)0, c, false, false);
}

// CDNA5 async global->LDS (bypasses VGPRs, tracked by ASYNCcnt)
__device__ __forceinline__ void async_ld128(uint32_t ldsAddr, uint64_t gaddr) {
    asm volatile("global_load_async_to_lds_b128 %0, %1, off"
                 :: "v"(ldsAddr), "v"(gaddr) : "memory");
}
__device__ __forceinline__ void wait_async0() {
    asm volatile("s_wait_asynccnt 0" ::: "memory");
}

// issue one 32-byte (2 x b128) per-thread chunk
__device__ __forceinline__ void issue_chunk(uint32_t ldsAddr, const u16* g) {
    async_ld128(ldsAddr,      (uint64_t)(uintptr_t)g);
    async_ld128(ldsAddr + 16, (uint64_t)(uintptr_t)(g + 8));
}

// ---------------- init / convert ----------------

__global__ void zero_counts_kernel(int* counts) {
    if (threadIdx.x < Ee) counts[threadIdx.x] = 0;
}

__global__ __launch_bounds__(256) void cvt_bf16_kernel(const float* __restrict__ in,
                                                       u16* __restrict__ out, long n) {
    long i = ((long)blockIdx.x * blockDim.x + threadIdx.x) * 4;
    if (i + 3 < n) {
        float4 v = *(const float4*)(in + i);
        out[i + 0] = f2bf(v.x);
        out[i + 1] = f2bf(v.y);
        out[i + 2] = f2bf(v.z);
        out[i + 3] = f2bf(v.w);
    }
}

// ---------------- gating: one wave per token ----------------

__global__ __launch_bounds__(128) void gate_kernel(const float* __restrict__ xt,
                                                   const float* __restrict__ Wg,
                                                   int* __restrict__ counts,
                                                   int* __restrict__ toklist,
                                                   float* __restrict__ wlist) {
    const int lane  = threadIdx.x & 31;
    const int token = blockIdx.x * 4 + (threadIdx.x >> 5);
    const float* x  = xt + (size_t)token * Dd;

    float acc[Ee];
#pragma unroll
    for (int e = 0; e < Ee; ++e) acc[e] = 0.f;

    for (int d = lane; d < Dd; d += 32) {
        float xv = x[d];
#pragma unroll
        for (int e = 0; e < Ee; ++e) acc[e] += xv * Wg[(size_t)e * Dd + d];
    }
#pragma unroll
    for (int e = 0; e < Ee; ++e) {
#pragma unroll
        for (int s = 16; s > 0; s >>= 1) acc[e] += __shfl_xor(acc[e], s, 32);
    }

    if (lane == 0) {
        float mx = acc[0];
#pragma unroll
        for (int e = 1; e < Ee; ++e) mx = fmaxf(mx, acc[e]);
        float sc[Ee]; float sum = 0.f;
#pragma unroll
        for (int e = 0; e < Ee; ++e) { sc[e] = __expf(acc[e] - mx); sum += sc[e]; }
        float inv = 1.f / sum;
#pragma unroll
        for (int e = 0; e < Ee; ++e) sc[e] *= inv;

        bool used[Ee];
#pragma unroll
        for (int e = 0; e < Ee; ++e) used[e] = false;

        for (int k = 0; k < KSEL; ++k) {
            int best = 0; float bv = -1.f;
#pragma unroll
            for (int e = 0; e < Ee; ++e)
                if (!used[e] && sc[e] > bv) { bv = sc[e]; best = e; }
            used[best] = true;
            int slot = atomicAdd(&counts[best], 1);
            toklist[(size_t)best * NTOK + slot] = token;
            wlist[(size_t)best * NTOK + slot]   = bv;   // ROUTE_SCALE == 1.0
        }
    }
}

__global__ void scan_kernel(const int* __restrict__ counts, int* __restrict__ offsets) {
    if (threadIdx.x == 0) {
        int o = 0;
        for (int e = 0; e < Ee; ++e) { offsets[e] = o; o += counts[e]; }
    }
}

// ---------------- double-buffered async WMMA mainloop (single B) ----------------
// One barrier per k-step:
//   s_wait_asynccnt 0  -> my writes to buf[cur] landed in LDS
//   s_barrier          -> all waves' buf[cur] writes visible AND everyone finished
//                         reading buf[cur^1] last iteration
//   issue buf[cur^1]   -> async DMA overlaps with the 8 WMMAs on buf[cur]

__device__ __forceinline__ void mm_loop_async(const u16* aRow, const u16* bRow,
                                              const u16* As, const u16* Bs,
                                              uint32_t daA, uint32_t daB,
                                              int Kdim, int lane, int wm, int wn,
                                              v8f acc[4][2]) {
    const int nk = Kdim / BK;
    issue_chunk(daA, aRow);
    issue_chunk(daB, bRow);
    for (int k = 0; k < nk; ++k) {
        const int cur = k & 1;
        wait_async0();
        __syncthreads();
        if (k + 1 < nk) {
            const uint32_t nb = (cur ^ 1) ? TILE_BYTES : 0u;
            issue_chunk(daA + nb, aRow + (size_t)(k + 1) * BK);
            issue_chunk(daB + nb, bRow + (size_t)(k + 1) * BK);
        }
        const u16* Ac = As + (size_t)cur * TILE_ELEMS;
        const u16* Bc = Bs + (size_t)cur * TILE_ELEMS;
        v16bf af[4];
#pragma unroll
        for (int mi = 0; mi < 4; ++mi) af[mi] = load_a(Ac, wm * 64 + mi * 16, lane);
#pragma unroll
        for (int ni = 0; ni < 2; ++ni) {
            v16bf bf = load_b(Bc, wn * 32 + ni * 16, lane);
#pragma unroll
            for (int mi = 0; mi < 4; ++mi) acc[mi][ni] = wmma_bf16(af[mi], bf, acc[mi][ni]);
        }
    }
    __syncthreads();
}

// ---------------- shared expert: s = bf16(silu(x @ Ws1^T + bs1)) ----------------

__global__ __launch_bounds__(256) void shared1_kernel(const u16* __restrict__ xtb,
                                                      const u16* __restrict__ ws1b,
                                                      const float* __restrict__ bs1,
                                                      u16* __restrict__ sbuf) {
    __shared__ u16 As[2 * TILE_ELEMS];
    __shared__ u16 Bs[2 * TILE_ELEMS];
    const int tid = threadIdx.x;
    const int m0 = blockIdx.x * BM, n0 = blockIdx.y * BN;
    const int lane = tid & 31, wm = (tid >> 5) & 1, wn = (tid >> 5) >> 1;
    const int r = tid >> 1, half = tid & 1;

    const u16* aRow = xtb + (size_t)(m0 + r) * Dd + half * 16;
    const u16* bRow = ws1b + (size_t)(n0 + r) * Dd + half * 16;
    const uint32_t daA = (uint32_t)(uintptr_t)&As[(size_t)r * LDT + half * 16];
    const uint32_t daB = (uint32_t)(uintptr_t)&Bs[(size_t)r * LDT + half * 16];

    v8f acc[4][2] = {};
    mm_loop_async(aRow, bRow, As, Bs, daA, daB, Dd, lane, wm, wn, acc);

    const int l15 = lane & 15, lhi = (lane >> 4) & 1;
#pragma unroll
    for (int ni = 0; ni < 2; ++ni) {
        const int n = n0 + wn * 32 + ni * 16 + l15;
        const float bb = bs1[n];
#pragma unroll
        for (int mi = 0; mi < 4; ++mi) {
#pragma unroll
            for (int rr = 0; rr < 8; ++rr) {
                const int m = m0 + wm * 64 + mi * 16 + rr + 8 * lhi;
                float v = acc[mi][ni][rr] + bb;
                float s = v * (1.f / (1.f + __expf(-v)));
                sbuf[(size_t)m * SHd + n] = f2bf(s);
            }
        }
    }
}

// ---------------- shared expert: out = s @ Ws2^T + bs2 (initializes out) ----------------

__global__ __launch_bounds__(256) void shared2_kernel(const u16* __restrict__ sbuf,
                                                      const u16* __restrict__ ws2b,
                                                      const float* __restrict__ bs2,
                                                      float* __restrict__ out) {
    __shared__ u16 As[2 * TILE_ELEMS];
    __shared__ u16 Bs[2 * TILE_ELEMS];
    const int tid = threadIdx.x;
    const int m0 = blockIdx.x * BM, n0 = blockIdx.y * BN;
    const int lane = tid & 31, wm = (tid >> 5) & 1, wn = (tid >> 5) >> 1;
    const int r = tid >> 1, half = tid & 1;

    const u16* aRow = sbuf + (size_t)(m0 + r) * SHd + half * 16;
    const u16* bRow = ws2b + (size_t)(n0 + r) * SHd + half * 16;
    const uint32_t daA = (uint32_t)(uintptr_t)&As[(size_t)r * LDT + half * 16];
    const uint32_t daB = (uint32_t)(uintptr_t)&Bs[(size_t)r * LDT + half * 16];

    v8f acc[4][2] = {};
    mm_loop_async(aRow, bRow, As, Bs, daA, daB, SHd, lane, wm, wn, acc);

    const int l15 = lane & 15, lhi = (lane >> 4) & 1;
#pragma unroll
    for (int ni = 0; ni < 2; ++ni) {
        const int n = n0 + wn * 32 + ni * 16 + l15;
        const float bb = bs2[n];
#pragma unroll
        for (int mi = 0; mi < 4; ++mi) {
#pragma unroll
            for (int rr = 0; rr < 8; ++rr) {
                const int m = m0 + wm * 64 + mi * 16 + rr + 8 * lhi;
                out[(size_t)m * Dd + n] = acc[mi][ni][rr] + bb;
            }
        }
    }
}

// ---------------- routed expert: h = silu(x@W1^T+b1) * (x@W3^T+b3) ----------------

__global__ __launch_bounds__(256) void expert_gemm1_kernel(const u16* __restrict__ xtb,
                                                           const u16* __restrict__ w1b,
                                                           const u16* __restrict__ w3b,
                                                           const float* __restrict__ b1,
                                                           const float* __restrict__ b3,
                                                           const int* __restrict__ counts,
                                                           const int* __restrict__ offsets,
                                                           const int* __restrict__ toklist,
                                                           u16* __restrict__ hbuf) {
    const int e   = blockIdx.z;
    const int cnt = counts[e];
    const int m0  = blockIdx.x * BM;
    if (m0 >= cnt) return;
    const int n0  = blockIdx.y * BN;
    const int off = offsets[e];

    __shared__ u16 As [2 * TILE_ELEMS];
    __shared__ u16 Bs1[2 * TILE_ELEMS];
    __shared__ u16 Bs3[2 * TILE_ELEMS];
    __shared__ int srcRow[BM];

    const int tid = threadIdx.x;
    if (tid < BM) {
        int m = m0 + tid; if (m > cnt - 1) m = cnt - 1;
        srcRow[tid] = toklist[(size_t)e * NTOK + m];
    }
    __syncthreads();

    const int lane = tid & 31, wm = (tid >> 5) & 1, wn = (tid >> 5) >> 1;
    const int r = tid >> 1, half = tid & 1;

    const u16* aRow = xtb + (size_t)srcRow[r] * Dd + half * 16;
    const u16* b1Row = w1b + ((size_t)e * Hh + n0 + r) * Dd + half * 16;
    const u16* b3Row = w3b + ((size_t)e * Hh + n0 + r) * Dd + half * 16;
    const uint32_t daA  = (uint32_t)(uintptr_t)&As [(size_t)r * LDT + half * 16];
    const uint32_t daB1 = (uint32_t)(uintptr_t)&Bs1[(size_t)r * LDT + half * 16];
    const uint32_t daB3 = (uint32_t)(uintptr_t)&Bs3[(size_t)r * LDT + half * 16];

    v8f acc1[4][2] = {};
    v8f acc3[4][2] = {};

    const int nk = Dd / BK;
    issue_chunk(daA, aRow);
    issue_chunk(daB1, b1Row);
    issue_chunk(daB3, b3Row);
    for (int k = 0; k < nk; ++k) {
        const int cur = k & 1;
        wait_async0();
        __syncthreads();
        if (k + 1 < nk) {
            const uint32_t nb = (cur ^ 1) ? TILE_BYTES : 0u;
            issue_chunk(daA + nb,  aRow  + (size_t)(k + 1) * BK);
            issue_chunk(daB1 + nb, b1Row + (size_t)(k + 1) * BK);
            issue_chunk(daB3 + nb, b3Row + (size_t)(k + 1) * BK);
        }
        const u16* Ac  = As  + (size_t)cur * TILE_ELEMS;
        const u16* B1c = Bs1 + (size_t)cur * TILE_ELEMS;
        const u16* B3c = Bs3 + (size_t)cur * TILE_ELEMS;
        v16bf af[4];
#pragma unroll
        for (int mi = 0; mi < 4; ++mi) af[mi] = load_a(Ac, wm * 64 + mi * 16, lane);
#pragma unroll
        for (int ni = 0; ni < 2; ++ni) {
            v16bf bf1 = load_b(B1c, wn * 32 + ni * 16, lane);
            v16bf bf3 = load_b(B3c, wn * 32 + ni * 16, lane);
#pragma unroll
            for (int mi = 0; mi < 4; ++mi) {
                acc1[mi][ni] = wmma_bf16(af[mi], bf1, acc1[mi][ni]);
                acc3[mi][ni] = wmma_bf16(af[mi], bf3, acc3[mi][ni]);
            }
        }
    }

    const int l15 = lane & 15, lhi = (lane >> 4) & 1;
#pragma unroll
    for (int ni = 0; ni < 2; ++ni) {
        const int n = n0 + wn * 32 + ni * 16 + l15;
        const float bb1 = b1[(size_t)e * Hh + n];
        const float bb3 = b3[(size_t)e * Hh + n];
#pragma unroll
        for (int mi = 0; mi < 4; ++mi) {
#pragma unroll
            for (int rr = 0; rr < 8; ++rr) {
                const int ml = wm * 64 + mi * 16 + rr + 8 * lhi;
                const int m  = m0 + ml;
                if (m < cnt) {
                    float v1 = acc1[mi][ni][rr] + bb1;
                    float v3 = acc3[mi][ni][rr] + bb3;
                    float hv = v1 * (1.f / (1.f + __expf(-v1))) * v3;
                    hbuf[(size_t)(off + m) * Hh + n] = f2bf(hv);
                }
            }
        }
    }
}

// ---------------- routed expert: out += (h @ W2^T + b2) * w ----------------

__global__ __launch_bounds__(256) void expert_gemm2_kernel(const u16* __restrict__ hbuf,
                                                           const u16* __restrict__ w2b,
                                                           const float* __restrict__ b2,
                                                           const int* __restrict__ counts,
                                                           const int* __restrict__ offsets,
                                                           const int* __restrict__ toklist,
                                                           const float* __restrict__ wlist,
                                                           float* __restrict__ out) {
    const int e   = blockIdx.z;
    const int cnt = counts[e];
    const int m0  = blockIdx.x * BM;
    if (m0 >= cnt) return;
    const int n0  = blockIdx.y * BN;
    const int off = offsets[e];

    __shared__ u16 As[2 * TILE_ELEMS];
    __shared__ u16 Bs[2 * TILE_ELEMS];
    __shared__ int srcRow[BM];
    __shared__ int tokS[BM];
    __shared__ float wS[BM];

    const int tid = threadIdx.x;
    if (tid < BM) {
        int m = m0 + tid; if (m > cnt - 1) m = cnt - 1;
        srcRow[tid] = off + m;
        tokS[tid]   = toklist[(size_t)e * NTOK + m];
        wS[tid]     = wlist[(size_t)e * NTOK + m];
    }
    __syncthreads();

    const int lane = tid & 31, wm = (tid >> 5) & 1, wn = (tid >> 5) >> 1;
    const int r = tid >> 1, half = tid & 1;

    const u16* aRow = hbuf + (size_t)srcRow[r] * Hh + half * 16;
    const u16* bRow = w2b + ((size_t)e * Dd + n0 + r) * Hh + half * 16;
    const uint32_t daA = (uint32_t)(uintptr_t)&As[(size_t)r * LDT + half * 16];
    const uint32_t daB = (uint32_t)(uintptr_t)&Bs[(size_t)r * LDT + half * 16];

    v8f acc[4][2] = {};
    mm_loop_async(aRow, bRow, As, Bs, daA, daB, Hh, lane, wm, wn, acc);

    const int l15 = lane & 15, lhi = (lane >> 4) & 1;
#pragma unroll
    for (int ni = 0; ni < 2; ++ni) {
        const int n = n0 + wn * 32 + ni * 16 + l15;
        const float bb = b2[(size_t)e * Dd + n];
#pragma unroll
        for (int mi = 0; mi < 4; ++mi) {
#pragma unroll
            for (int rr = 0; rr < 8; ++rr) {
                const int ml = wm * 64 + mi * 16 + rr + 8 * lhi;
                if (m0 + ml < cnt) {
                    float v = (acc[mi][ni][rr] + bb) * wS[ml];
                    atomicAdd(out + (size_t)tokS[ml] * Dd + n, v);
                }
            }
        }
    }
}

// ---------------- host launch ----------------

extern "C" void kernel_launch(void* const* d_in, const int* in_sizes, int n_in,
                              void* d_out, int out_size, void* d_ws, size_t ws_size,
                              hipStream_t stream) {
    const float* emb = (const float*)d_in[0];   // [B,S,C,D] == xt [N, D]
    const float* Wg  = (const float*)d_in[2];
    const float* W1  = (const float*)d_in[3];
    const float* b1  = (const float*)d_in[4];
    const float* W2  = (const float*)d_in[5];
    const float* b2  = (const float*)d_in[6];
    const float* W3  = (const float*)d_in[7];
    const float* b3  = (const float*)d_in[8];
    const float* Ws1 = (const float*)d_in[9];
    const float* bs1 = (const float*)d_in[10];
    const float* Ws2 = (const float*)d_in[11];
    const float* bs2 = (const float*)d_in[12];
    float* out = (float*)d_out;

    char* p = (char*)d_ws;
    auto alloc = [&](size_t bytes) -> void* {
        void* r = (void*)p;
        p += (bytes + 255) & ~(size_t)255;
        return r;
    };

    int*   counts  = (int*)alloc(Ee * sizeof(int));
    int*   offsets = (int*)alloc(Ee * sizeof(int));
    int*   toklist = (int*)alloc((size_t)Ee * NTOK * sizeof(int));
    float* wlist   = (float*)alloc((size_t)Ee * NTOK * sizeof(float));
    u16*   xtb     = (u16*)alloc((size_t)NTOK * Dd * 2);
    u16*   w1b     = (u16*)alloc((size_t)Ee * Hh * Dd * 2);
    u16*   w3b     = (u16*)alloc((size_t)Ee * Hh * Dd * 2);
    u16*   w2b     = (u16*)alloc((size_t)Ee * Dd * Hh * 2);
    u16*   ws1b    = (u16*)alloc((size_t)SHd * Dd * 2);
    u16*   ws2b    = (u16*)alloc((size_t)Dd * SHd * 2);
    u16*   sbuf    = (u16*)alloc((size_t)NTOK * SHd * 2);
    u16*   hbuf    = (u16*)alloc((size_t)NTOK * KSEL * Hh * 2);

    zero_counts_kernel<<<1, 32, 0, stream>>>(counts);

    auto cvt = [&](const float* src, u16* dst, long n) {
        long blocks = (n / 4 + 255) / 256;
        cvt_bf16_kernel<<<(int)blocks, 256, 0, stream>>>(src, dst, n);
    };
    cvt(emb, xtb, (long)NTOK * Dd);
    cvt(W1,  w1b, (long)Ee * Hh * Dd);
    cvt(W3,  w3b, (long)Ee * Hh * Dd);
    cvt(W2,  w2b, (long)Ee * Dd * Hh);
    cvt(Ws1, ws1b, (long)SHd * Dd);
    cvt(Ws2, ws2b, (long)Dd * SHd);

    gate_kernel<<<NTOK / 4, 128, 0, stream>>>(emb, Wg, counts, toklist, wlist);
    scan_kernel<<<1, 32, 0, stream>>>(counts, offsets);

    shared1_kernel<<<dim3(NTOK / BM, SHd / BN), 256, 0, stream>>>(xtb, ws1b, bs1, sbuf);
    shared2_kernel<<<dim3(NTOK / BM, Dd / BN), 256, 0, stream>>>(sbuf, ws2b, bs2, out);

    expert_gemm1_kernel<<<dim3(NTOK / BM, Hh / BN, Ee), 256, 0, stream>>>(
        xtb, w1b, w3b, b1, b3, counts, offsets, toklist, hbuf);
    expert_gemm2_kernel<<<dim3(NTOK / BM, Dd / BN, Ee), 256, 0, stream>>>(
        hbuf, w2b, b2, counts, offsets, toklist, wlist, out);
}
